// RecurrentMemory_49667001811618
// MI455X (gfx1250) — matrix operand order repaired and checked by reading, not compiled, not run on hardware
//
#include <hip/hip_runtime.h>
#include <math.h>

typedef __attribute__((ext_vector_type(16))) _Float16 v16h;
typedef __attribute__((ext_vector_type(8)))  _Float16 v8h;
typedef __attribute__((ext_vector_type(8)))  float    v8f;

#define HID 512
#define VOC 50257
#define MEM 200
#define BAT 64
#define LSCH 32          // log-softmax chunks per row
#define LSSPAN 1571      // ceil(VOC/LSCH); 32*1571 = 50272 >= VOC

__device__ __forceinline__ float sigf(float x) { return 1.0f / (1.0f + __expf(-x)); }

// ---------------------------------------------------------------------------
// Generic single-wave WMMA GEMM. A is pre-converted f16, row-major with
// leading dim lda (multiple of 8 -> 16B-aligned v8h loads). Each 32-thread
// block computes a 64x16 strip (4 row tiles share one B tile) so the fp32
// B matrix streams from HBM exactly once. A and B tiles both use contiguous k = 16*(lane/16) + i.
// ---------------------------------------------------------------------------
template <typename F>
__global__ void wmma_gemm(const _Float16* __restrict__ A, int lda, F f, int K) {
    const int lane = threadIdx.x;           // 0..31 (wave32)
    const int col0 = blockIdx.x * 16;
    const int m_in = lane & 15;
    const int klo  = (lane < 16) ? 0 : 16;  // A: VGPR0..3 K range
    const int khi  = (lane < 16) ? 8 : 24;  // A: VGPR4..7 K range
    const int ncol = lane & 15;             // B/D: column = lane%16
    const int kb   = (lane < 16) ? 0 : 16;  // B: K base per half-wave

    v8f zero = {};
    v8f acc[4];
    acc[0] = zero; acc[1] = zero; acc[2] = zero; acc[3] = zero;

    for (int k0 = 0; k0 < K; k0 += 32) {
        f.prefetch(col0 + ncol, k0);
        // B tile 32x16: lane holds column ncol, K = k0+kb+i (fp32 -> f16)
        v16h bt;
#pragma unroll
        for (int i = 0; i < 16; ++i)
            bt[i] = (_Float16)f.loadB(col0 + ncol, k0 + kb + i);
#pragma unroll
        for (int rt = 0; rt < 4; ++rt) {
            const _Float16* ap = A + (size_t)(rt * 16 + m_in) * lda + k0;
            v8h alo = *(const v8h*)(ap + klo);
            v8h ahi = *(const v8h*)(ap + khi);
            v16h at = __builtin_shufflevector(alo, ahi, 0, 1, 2, 3, 4, 5, 6, 7,
                                              8, 9, 10, 11, 12, 13, 14, 15);
            acc[rt] = __builtin_amdgcn_wmma_f32_16x16x32_f16(
                false, at, false, bt, (short)0, acc[rt], false, false);
        }
    }
    const int mofs = (lane < 16) ? 0 : 8;
#pragma unroll
    for (int rt = 0; rt < 4; ++rt)
#pragma unroll
        for (int r = 0; r < 8; ++r)
            f.store(rt * 16 + r + mofs, col0 + ncol, acc[rt][r]);
}

// ------------------------------- functors ----------------------------------
struct LstmW {                      // gates = A(1536) @ [Wih|Whh]^T + bih + bhh
    const float *Wih, *Whh, *bih, *bhh;
    float* gates;
    __device__ void prefetch(int, int) const {}
    __device__ float loadB(int col, int k) const {
        return (k < 2 * HID) ? Wih[(size_t)col * 2 * HID + k]
                             : Whh[(size_t)col * HID + (k - 2 * HID)];
    }
    __device__ void store(int row, int col, float v) const {
        gates[row * 4 * HID + col] = v + bih[col] + bhh[col];
    }
};

struct AttnW {                      // attn2 = scores16(224) @ attn_W^T + b
    const float *W, *bias;
    float* attn2;
    __device__ void prefetch(int, int) const {}
    __device__ float loadB(int col, int k) const {
        return (col < MEM && k < MEM) ? W[col * MEM + k] : 0.f;
    }
    __device__ void store(int row, int col, float v) const {
        if (col < MEM) attn2[row * MEM + col] = v + bias[col];
    }
};

struct GruW {                       // out = A(512) @ W^T + bias  (launched twice)
    const float *W, *bias;
    float* out;
    __device__ void prefetch(int, int) const {}
    __device__ float loadB(int col, int k) const { return W[(size_t)col * HID + k]; }
    __device__ void store(int row, int col, float v) const {
        out[row * 3 * HID + col] = v + bias[col];
    }
};

struct OutW {                       // logits = mbc16(1024) @ out_W^T + out_b
    const float *W, *bias;
    float* logits;
    __device__ void prefetch(int col, int k) const {
        if (col < VOC)   // speculative prefetch ~2KB ahead on this lane's row
            __builtin_prefetch(&W[(size_t)col * 2 * HID + k + 512], 0, 1);
    }
    __device__ float loadB(int col, int k) const {
        return (col < VOC) ? W[(size_t)col * 2 * HID + k] : 0.f;
    }
    __device__ void store(int row, int col, float v) const {
        if (col < VOC) logits[(size_t)row * VOC + col] = v + bias[col];
    }
};

// --------------------------- supporting kernels ----------------------------
// A for LSTM: f16 [relu(emb[ids]) | ctx | h0], B x 1536
__global__ void prep_lstm_a(const float* __restrict__ emb, const int* __restrict__ ids,
                            const float* __restrict__ ctx, const float* __restrict__ h0,
                            _Float16* __restrict__ a) {
    int t = blockIdx.x * blockDim.x + threadIdx.x;
    if (t >= BAT * 3 * HID) return;
    int b = t / (3 * HID), j = t % (3 * HID);
    float v;
    if (j < HID)            v = fmaxf(emb[(size_t)ids[b] * HID + j], 0.f);
    else if (j < 2 * HID)   v = ctx[b * HID + (j - HID)];
    else                    v = h0[b * HID + (j - 2 * HID)];
    a[t] = (_Float16)v;
}

__global__ void lstm_pointwise(const float* __restrict__ gates, const float* __restrict__ c0,
                               float* __restrict__ h1_ws, _Float16* __restrict__ h116,
                               float* __restrict__ h1_out, float* __restrict__ c1_out) {
    int t = blockIdx.x * blockDim.x + threadIdx.x;
    if (t >= BAT * HID) return;
    int b = t / HID, j = t % HID;
    const float* g = gates + b * 4 * HID;
    float i  = sigf(g[j]);
    float fg = sigf(g[HID + j]);
    float gg = tanhf(g[2 * HID + j]);
    float o  = sigf(g[3 * HID + j]);
    float c = fg * c0[t] + i * gg;
    float h = o * tanhf(c);
    h1_ws[t] = h; h116[t] = (_Float16)h; h1_out[t] = h; c1_out[t] = c;
}

// attn[b,m] = dot(m_emb[memory[m,b]], h1[b]) : one wave per (m,b) gather-dot
__global__ void attn_scores(const float* __restrict__ m_emb, const int* __restrict__ mem,
                            const float* __restrict__ h1, float* __restrict__ scores) {
    int wave = (blockIdx.x * blockDim.x + threadIdx.x) >> 5;
    int lane = threadIdx.x & 31;
    if (wave >= MEM * BAT) return;
    int m = wave / BAT, b = wave % BAT;
    const float* row = m_emb + (size_t)mem[m * BAT + b] * HID;
    const float* hv  = h1 + b * HID;
    float s = 0.f;
    for (int j = lane; j < HID; j += 32) s += row[j] * hv[j];
#pragma unroll
    for (int off = 16; off > 0; off >>= 1) s += __shfl_down(s, off, 32);
    if (lane == 0) scores[b * MEM + m] = s;
}

// f16 zero-padded A for attn GEMM: B x 224
__global__ void cvt_scores(const float* __restrict__ scores, _Float16* __restrict__ s16) {
    int t = blockIdx.x * blockDim.x + threadIdx.x;
    if (t >= BAT * 224) return;
    int b = t / 224, k = t % 224;
    s16[t] = (k < MEM) ? (_Float16)scores[b * MEM + k] : (_Float16)0.f;
}

// per-batch: log_softmax over M, then mctx16[b,h] = sum_m c_emb[mem[m,b]][h]*logp
__global__ void mem_context(const float* __restrict__ attn2, const float* __restrict__ c_emb,
                            const int* __restrict__ mem, _Float16* __restrict__ mctx16) {
    __shared__ float red[256];
    __shared__ float ldist[MEM];
    __shared__ int   rows[MEM];
    int b = blockIdx.x, tid = threadIdx.x;
    float mx = -INFINITY;
    for (int m = tid; m < MEM; m += 256) mx = fmaxf(mx, attn2[b * MEM + m]);
    red[tid] = mx; __syncthreads();
    for (int s = 128; s > 0; s >>= 1) { if (tid < s) red[tid] = fmaxf(red[tid], red[tid + s]); __syncthreads(); }
    mx = red[0]; __syncthreads();
    float se = 0.f;
    for (int m = tid; m < MEM; m += 256) se += __expf(attn2[b * MEM + m] - mx);
    red[tid] = se; __syncthreads();
    for (int s = 128; s > 0; s >>= 1) { if (tid < s) red[tid] += red[tid + s]; __syncthreads(); }
    float lse = mx + __logf(red[0]); __syncthreads();
    for (int m = tid; m < MEM; m += 256) { ldist[m] = attn2[b * MEM + m] - lse; rows[m] = mem[m * BAT + b]; }
    __syncthreads();
    for (int j = tid; j < HID; j += 256) {
        float acc = 0.f;
        for (int m = 0; m < MEM; ++m) acc += c_emb[(size_t)rows[m] * HID + j] * ldist[m];
        mctx16[b * HID + j] = (_Float16)acc;
    }
}

// GRU combine -> mbc16 = f16 [mb | ctx], B x 1024
__global__ void gru_pointwise(const float* __restrict__ gi, const float* __restrict__ gh,
                              const float* __restrict__ h1, const float* __restrict__ ctx,
                              _Float16* __restrict__ mbc16) {
    int t = blockIdx.x * blockDim.x + threadIdx.x;
    if (t >= BAT * HID) return;
    int b = t / HID, j = t % HID;
    const float* a = gi + b * 3 * HID;
    const float* c = gh + b * 3 * HID;
    float r = sigf(a[j] + c[j]);
    float z = sigf(a[HID + j] + c[HID + j]);
    float n = tanhf(a[2 * HID + j] + r * c[2 * HID + j]);
    float mb = (1.f - z) * n + z * h1[t];
    mbc16[b * 2 * HID + j]       = (_Float16)mb;
    mbc16[b * 2 * HID + HID + j] = (_Float16)ctx[t];
}

// ---- parallel log-softmax over V: partial (max,sumexp) -> combine -> subtract
__global__ void ls_partial(const float* __restrict__ logits, float* __restrict__ part) {
    __shared__ float red[256];
    int b = blockIdx.x, c = blockIdx.y, tid = threadIdx.x;
    int lo = c * LSSPAN, hi = min(lo + LSSPAN, VOC);
    const float* row = logits + (size_t)b * VOC;
    float mx = -INFINITY;
    for (int j = lo + tid; j < hi; j += 256) mx = fmaxf(mx, row[j]);
    red[tid] = mx; __syncthreads();
    for (int s = 128; s > 0; s >>= 1) { if (tid < s) red[tid] = fmaxf(red[tid], red[tid + s]); __syncthreads(); }
    mx = red[0]; __syncthreads();
    float se = 0.f;
    for (int j = lo + tid; j < hi; j += 256) se += __expf(row[j] - mx);
    red[tid] = se; __syncthreads();
    for (int s = 128; s > 0; s >>= 1) { if (tid < s) red[tid] += red[tid + s]; __syncthreads(); }
    if (tid == 0) { part[(b * LSCH + c) * 2] = mx; part[(b * LSCH + c) * 2 + 1] = red[0]; }
}

__global__ void ls_combine(const float* __restrict__ part, float* __restrict__ lse) {
    int b = blockIdx.x, lane = threadIdx.x;          // 32 lanes == LSCH chunks
    float m = part[(b * LSCH + lane) * 2];
    float s = part[(b * LSCH + lane) * 2 + 1];
    float M = m;
#pragma unroll
    for (int off = 16; off > 0; off >>= 1) M = fmaxf(M, __shfl_xor(M, off, 32));
    float sr = s * __expf(m - M);
#pragma unroll
    for (int off = 16; off > 0; off >>= 1) sr += __shfl_xor(sr, off, 32);
    if (lane == 0) lse[b] = M + __logf(sr);
}

__global__ void ls_subtract(float* __restrict__ logits, const float* __restrict__ lse) {
    int b = blockIdx.y;
    int j = blockIdx.x * blockDim.x + threadIdx.x;
    if (j < VOC) logits[(size_t)b * VOC + j] -= lse[b];
}

// ---------------------------------------------------------------------------
extern "C" void kernel_launch(void* const* d_in, const int* in_sizes, int n_in,
                              void* d_out, int out_size, void* d_ws, size_t ws_size,
                              hipStream_t stream) {
    const float* rmn_hidden = (const float*)d_in[0];
    const float* cell       = (const float*)d_in[1];
    const float* context    = (const float*)d_in[2];
    const float* emb        = (const float*)d_in[3];
    const float* m_emb      = (const float*)d_in[4];
    const float* c_emb      = (const float*)d_in[5];
    const float* lstm_W_ih  = (const float*)d_in[6];
    const float* lstm_W_hh  = (const float*)d_in[7];
    const float* lstm_b_ih  = (const float*)d_in[8];
    const float* lstm_b_hh  = (const float*)d_in[9];
    const float* gru_W_ih   = (const float*)d_in[10];
    const float* gru_W_hh   = (const float*)d_in[11];
    const float* gru_b_ih   = (const float*)d_in[12];
    const float* gru_b_hh   = (const float*)d_in[13];
    const float* attn_W     = (const float*)d_in[14];
    const float* attn_b     = (const float*)d_in[15];
    const float* out_W      = (const float*)d_in[16];
    const float* out_b      = (const float*)d_in[17];
    const int*   input_ids  = (const int*)d_in[18];
    const int*   memory_t   = (const int*)d_in[19];

    // workspace carve-up (all segment sizes multiples of 4 floats -> 16B aligned)
    float* ws       = (float*)d_ws;
    float* gates    = ws;                          // B*4H          = 131072 f
    float* h1       = gates + BAT * 4 * HID;       // B*H           = 32768 f
    float* scores   = h1 + BAT * HID;              // B*M           = 12800 f
    float* attn2    = scores + BAT * MEM;          // B*M           = 12800 f
    float* gi       = attn2 + BAT * MEM;           // B*3H          = 98304 f
    float* gh       = gi + BAT * 3 * HID;          // B*3H          = 98304 f
    float* part     = gh + BAT * 3 * HID;          // B*LSCH*2      = 4096 f
    float* lse      = part + BAT * LSCH * 2;       // B             = 64 f
    _Float16* a_lstm  = (_Float16*)(lse + 64);     // B*3H h        = 49152 f
    _Float16* h116    = a_lstm + BAT * 3 * HID;    // B*H h         = 16384 f
    _Float16* scores16= h116 + BAT * HID;          // B*224 h       = 7168 f
    _Float16* mctx16  = scores16 + BAT * 224;      // B*H h         = 16384 f
    _Float16* mbc16   = mctx16 + BAT * HID;        // B*2H h        = 32768 f

    float* out    = (float*)d_out;
    float* logits = out;                           // (B,V)
    float* h1_out = out + (size_t)BAT * VOC;       // (B,H)
    float* c1_out = h1_out + (size_t)BAT * HID;    // (B,H)

    // 1) f16 A for LSTM: [relu(emb[ids]) | ctx | h0]
    prep_lstm_a<<<(BAT * 3 * HID + 255) / 256, 256, 0, stream>>>(
        emb, input_ids, context, rmn_hidden, a_lstm);

    // 2) LSTM gates via WMMA: (B x 4H), K = 1536
    LstmW lw{lstm_W_ih, lstm_W_hh, lstm_b_ih, lstm_b_hh, gates};
    wmma_gemm<LstmW><<<(4 * HID) / 16, 32, 0, stream>>>(a_lstm, 3 * HID, lw, 3 * HID);

    // 3) pointwise LSTM -> h1 (f32 + f16), c1
    lstm_pointwise<<<(BAT * HID + 255) / 256, 256, 0, stream>>>(
        gates, cell, h1, h116, h1_out, c1_out);

    // 4) attention scores (gather-dot)
    attn_scores<<<(MEM * BAT) / 8, 256, 0, stream>>>(m_emb, memory_t, h1, scores);

    // 5) f16 zero-padded scores, then attn2 = scores @ attn_W^T + b via WMMA
    cvt_scores<<<(BAT * 224 + 255) / 256, 256, 0, stream>>>(scores, scores16);
    AttnW aw{attn_W, attn_b, attn2};
    wmma_gemm<AttnW><<<(MEM + 15) / 16, 32, 0, stream>>>(scores16, 224, aw, 224);

    // 6) log_softmax over M + gathered memory context (f16)
    mem_context<<<BAT, 256, 0, stream>>>(attn2, c_emb, memory_t, mctx16);

    // 7) GRU gi/gh via WMMA: (B x 3H), K = H
    GruW gwi{gru_W_ih, gru_b_ih, gi};
    wmma_gemm<GruW><<<(3 * HID) / 16, 32, 0, stream>>>(mctx16, HID, gwi, HID);
    GruW gwh{gru_W_hh, gru_b_hh, gh};
    wmma_gemm<GruW><<<(3 * HID) / 16, 32, 0, stream>>>(h116, HID, gwh, HID);

    // 8) pointwise GRU -> mbc16 = [mb | ctx]
    gru_pointwise<<<(BAT * HID + 255) / 256, 256, 0, stream>>>(gi, gh, h1, context, mbc16);

    // 9) big projection: logits = mbc @ out_W^T + out_b (weights streamed once)
    OutW ow{out_W, out_b, logits};
    wmma_gemm<OutW><<<(VOC + 15) / 16, 32, 0, stream>>>(mbc16, 2 * HID, ow, 2 * HID);

    // 10) parallel log-softmax over V
    ls_partial<<<dim3(BAT, LSCH), 256, 0, stream>>>(logits, part);
    ls_combine<<<BAT, 32, 0, stream>>>(part, lse);
    ls_subtract<<<dim3((VOC + 255) / 256, BAT), 256, 0, stream>>>(logits, lse);
}
